// LoRAFFN_70695161692198
// MI455X (gfx1250) — matrix operand order repaired and checked by reading, not compiled, not run on hardware
//
#include <hip/hip_runtime.h>

// ---------------------------------------------------------------------------
// LoRA FFN on MI455X (gfx1250):
//   h   = gelu(x @ W1^T + b1)
//   out = h @ (W2 + SCALE*(A@B_lora))^T + b2      (LoRA folded into W2eff)
// Two bf16 WMMA GEMMs (v_wmma_f32_16x16x32_bf16), fp32 accumulate.
// BK=64, double-buffered LDS, async global->LDS staging (ASYNCcnt path).
// ---------------------------------------------------------------------------

#define TRY_ASYNC 1   // inline-asm global_load_async_to_lds_b128 staging

typedef __attribute__((ext_vector_type(16))) __bf16 v16bf;
typedef __attribute__((ext_vector_type(2)))  __bf16 v2bf;
typedef __attribute__((ext_vector_type(8)))  float  v8f;
typedef __attribute__((ext_vector_type(4)))  float  v4f;
typedef __attribute__((ext_vector_type(4)))  unsigned int v4u;
typedef __attribute__((ext_vector_type(2)))  unsigned int v2u;

union FragBF { v4u u[2]; v16bf v; };

constexpr int   D      = 2048;            // D_MODEL == D_FF == K == N
constexpr int   M_TOT  = 4 * 4096;        // B_SZ * SEQ
constexpr float SCALE  = 1.0f / 4.0f;     // ALPHA / RANK
constexpr int   BM     = 128;
constexpr int   BN     = 128;
constexpr int   BK     = 64;
constexpr int   LDPAD  = BK + 8;          // padded LDS row stride (elements)

__device__ __forceinline__ unsigned short f2bf(float x) {
  unsigned int u = __float_as_uint(x);
  u += 0x7FFFu + ((u >> 16) & 1u);        // round-to-nearest-even
  return (unsigned short)(u >> 16);
}

__device__ __forceinline__ unsigned int pack2(float a, float b) {
#if __has_builtin(__builtin_amdgcn_cvt_pk_bf16_f32)
  union { v2bf v; unsigned int u; } c;
  c.v = __builtin_amdgcn_cvt_pk_bf16_f32(a, b);
  return c.u;
#else
  return (unsigned int)f2bf(a) | ((unsigned int)f2bf(b) << 16);
#endif
}

__device__ __forceinline__ float fast_tanh(float x) {
#if __has_builtin(__builtin_amdgcn_tanhf)
  return __builtin_amdgcn_tanhf(x);
#elif __has_builtin(__builtin_amdgcn_tanh_f32)
  return __builtin_amdgcn_tanh_f32(x);
#else
  return tanhf(x);
#endif
}

__device__ __forceinline__ float gelu_tanh(float x) {
  const float c = 0.7978845608028654f;    // sqrt(2/pi)
  float t = fast_tanh(c * (x + 0.044715f * x * x * x));
  return 0.5f * x * (1.0f + t);
}

// 16-byte global -> LDS copy. Async variant uses the CDNA5 TDM-adjacent
// GLOBAL_LOAD_ASYNC_TO_LDS_B128 (tracked by ASYNCcnt), per-lane addresses.
__device__ __forceinline__ void copy16(unsigned short* l, const unsigned short* g) {
#if TRY_ASYNC
  unsigned int lo = (unsigned int)(unsigned long long)l;  // low 32b = LDS offset
  asm volatile("global_load_async_to_lds_b128 %0, %1, off"
               :: "v"(lo), "v"(g) : "memory");
#else
  *reinterpret_cast<v4u*>(l) = *reinterpret_cast<const v4u*>(g);
#endif
}

__device__ __forceinline__ void async_fence() {
#if TRY_ASYNC
  asm volatile("s_wait_asynccnt 0x0" ::: "memory");
#endif
}

// ---- W1 (fp32) -> bf16, 8 elements / thread ------------------------------
__global__ __launch_bounds__(256) void prep_w1_kernel(
    const float* __restrict__ W1, unsigned short* __restrict__ W1b) {
  size_t i = ((size_t)blockIdx.x * 256 + threadIdx.x) * 8;
  v4f f0 = *reinterpret_cast<const v4f*>(W1 + i);
  v4f f1 = *reinterpret_cast<const v4f*>(W1 + i + 4);
  v4u o = { pack2(f0.x, f0.y), pack2(f0.z, f0.w),
            pack2(f1.x, f1.y), pack2(f1.z, f1.w) };
  *reinterpret_cast<v4u*>(W1b + i) = o;
}

// ---- W2eff = W2 + SCALE * (A @ B_lora), -> bf16, 4 elems / thread --------
__global__ __launch_bounds__(256) void prep_w2_kernel(
    const float* __restrict__ W2, const float* __restrict__ A,
    const float* __restrict__ Bl, unsigned short* __restrict__ W2b) {
  size_t t = (size_t)blockIdx.x * 256 + threadIdx.x;
  int o = (int)(t >> 9);                  // row  (D_MODEL), 512 threads per row
  int f = (int)(t & 511) << 2;            // col  (D_FF), 4 per thread
  float a0 = A[o * 4 + 0] * SCALE;
  float a1 = A[o * 4 + 1] * SCALE;
  float a2 = A[o * 4 + 2] * SCALE;
  float a3 = A[o * 4 + 3] * SCALE;
  v4f w  = *reinterpret_cast<const v4f*>(W2 + (size_t)o * D + f);
  v4f r0 = *reinterpret_cast<const v4f*>(Bl + 0 * D + f);
  v4f r1 = *reinterpret_cast<const v4f*>(Bl + 1 * D + f);
  v4f r2 = *reinterpret_cast<const v4f*>(Bl + 2 * D + f);
  v4f r3 = *reinterpret_cast<const v4f*>(Bl + 3 * D + f);
  float e0 = w.x + a0 * r0.x + a1 * r1.x + a2 * r2.x + a3 * r3.x;
  float e1 = w.y + a0 * r0.y + a1 * r1.y + a2 * r2.y + a3 * r3.y;
  float e2 = w.z + a0 * r0.z + a1 * r1.z + a2 * r2.z + a3 * r3.z;
  float e3 = w.w + a0 * r0.w + a1 * r1.w + a2 * r2.w + a3 * r3.w;
  v2u out = { pack2(e0, e1), pack2(e2, e3) };
  *reinterpret_cast<v2u*>(W2b + (size_t)o * D + f) = out;
}

// ---- Tiled bf16 WMMA GEMM:  C[M,N] = act(A[M,K] * Bw[N,K]^T + bias) ------
// Block: 256 thr = 8 waves (2x4). WG tile 128x128, wave tile 64x32, BK=64.
// Double-buffered LDS, one barrier per K-step, 16 WMMAs / step / wave.
template <bool A_F32, bool DO_GELU>
__global__ __launch_bounds__(256) void gemm_bf16_wmma(
    const void* __restrict__ Ap, const unsigned short* __restrict__ Bw,
    const float* __restrict__ bias, void* __restrict__ Out) {
  __shared__ unsigned short As[2][BM][LDPAD];
  __shared__ unsigned short Bs[2][BN][LDPAD];

  const int tid  = threadIdx.x;
  const int lane = tid & 31;
  const int wave = tid >> 5;
  const int wm   = wave >> 2;             // 0..1  (M position of wave)
  const int wn   = wave & 3;              // 0..3  (N position of wave)
  const int half = lane >> 4;             // lane group 0/1
  const int l15  = lane & 15;

  const int    gn0 = blockIdx.x * BN;
  const size_t gm0 = (size_t)blockIdx.y * BM;

  v8f acc[4][2];
  const v8f vzero = {0.f, 0.f, 0.f, 0.f, 0.f, 0.f, 0.f, 0.f};
#pragma unroll
  for (int i = 0; i < 4; ++i)
#pragma unroll
    for (int j = 0; j < 2; ++j) acc[i][j] = vzero;

  const int srow = tid >> 1;              // 0..127 : staged tile row
  const int sseg = (tid & 1) * 32;        // element offset within 64-wide row

  // stage one BK-tile (A + B) into LDS buffer `buf` at K offset k0
  auto stage = [&](int buf, int k0) {
    if constexpr (A_F32) {                // x fp32 -> bf16 while staging
      const float* src = (const float*)Ap + (gm0 + srow) * D + k0 + sseg;
      unsigned int p[16];
#pragma unroll
      for (int q = 0; q < 8; ++q) {
        v4f f = *reinterpret_cast<const v4f*>(src + q * 4);
        p[2 * q]     = pack2(f.x, f.y);
        p[2 * q + 1] = pack2(f.z, f.w);
      }
#pragma unroll
      for (int q = 0; q < 4; ++q)
        *reinterpret_cast<v4u*>(&As[buf][srow][sseg + 8 * q]) =
            (v4u){p[4 * q], p[4 * q + 1], p[4 * q + 2], p[4 * q + 3]};
    } else {                              // h bf16: async direct to LDS
      const unsigned short* src =
          (const unsigned short*)Ap + (gm0 + srow) * D + k0 + sseg;
#pragma unroll
      for (int q = 0; q < 4; ++q)
        copy16(&As[buf][srow][sseg + 8 * q], src + 8 * q);
    }
    const unsigned short* srcB = Bw + ((size_t)gn0 + srow) * D + k0 + sseg;
#pragma unroll
    for (int q = 0; q < 4; ++q)
      copy16(&Bs[buf][srow][sseg + 8 * q], srcB + 8 * q);
    __builtin_prefetch(srcB + BK, 0, 1);  // global_prefetch one tile ahead
  };

  // 16 WMMAs on LDS buffer `buf` (two K=32 sub-steps)
  auto mac = [&](int buf) {
#pragma unroll
    for (int kk = 0; kk < BK; kk += 32) {
      v16bf afr[4], bfr[2];
#pragma unroll
      for (int i = 0; i < 4; ++i) {        // A 16x32: K halves interleaved
        const int r = wm * 64 + i * 16 + l15;
        FragBF fa;
        fa.u[0] = *reinterpret_cast<const v4u*>(&As[buf][r][kk + half * 8]);
        fa.u[1] = *reinterpret_cast<const v4u*>(&As[buf][r][kk + 16 + half * 8]);
        afr[i] = fa.v;
      }
#pragma unroll
      for (int j = 0; j < 2; ++j) {        // B 32x16: K halves interleaved
        const int r = wn * 32 + j * 16 + l15;
        FragBF fb;
        fb.u[0] = *reinterpret_cast<const v4u*>(&Bs[buf][r][kk + half * 8]);
        fb.u[1] = *reinterpret_cast<const v4u*>(&Bs[buf][r][kk + 16 + half * 8]);
        bfr[j] = fb.v;
      }
#pragma unroll
      for (int i = 0; i < 4; ++i)
#pragma unroll
        for (int j = 0; j < 2; ++j)
          acc[i][j] = __builtin_amdgcn_wmma_f32_16x16x32_bf16(
              false, afr[i], false, bfr[j], (short)0, acc[i][j], false, false);
    }
  };

  stage(0, 0);                            // prologue
  int cur = 0;
  for (int k0 = 0; k0 < D; k0 += BK) {
    async_fence();                        // our async LDS writes have landed
    __syncthreads();                      // tile `cur` visible to all waves
    if (k0 + BK < D) stage(cur ^ 1, k0 + BK);
    mac(cur);
    cur ^= 1;
  }

  // ---- epilogue ----
#pragma unroll
  for (int j = 0; j < 2; ++j) {
    const int col = gn0 + wn * 32 + j * 16 + l15;
    const float bv = bias[col];
#pragma unroll
    for (int i = 0; i < 4; ++i) {
#pragma unroll
      for (int v = 0; v < 8; ++v) {
        const size_t row = gm0 + wm * 64 + i * 16 + v + 8 * half;
        float val = acc[i][j][v] + bv;
        if constexpr (DO_GELU) {
          ((unsigned short*)Out)[row * D + col] = f2bf(gelu_tanh(val));
        } else {
          ((float*)Out)[row * D + col] = val;
        }
      }
    }
  }
}

extern "C" void kernel_launch(void* const* d_in, const int* in_sizes, int n_in,
                              void* d_out, int out_size, void* d_ws, size_t ws_size,
                              hipStream_t stream) {
  const float* x  = (const float*)d_in[0];   // [4,4096,2048]
  const float* W1 = (const float*)d_in[1];   // [2048,2048]
  const float* b1 = (const float*)d_in[2];   // [2048]
  const float* W2 = (const float*)d_in[3];   // [2048,2048]
  const float* b2 = (const float*)d_in[4];   // [2048]
  const float* A  = (const float*)d_in[5];   // [2048,4]
  const float* Bl = (const float*)d_in[6];   // [4,2048]

  // workspace layout (bf16 buffers): W1b | W2b | h   (~84 MB total)
  unsigned short* W1b = (unsigned short*)d_ws;
  unsigned short* W2b = W1b + (size_t)D * D;
  unsigned short* h   = W2b + (size_t)D * D;

  // weight prep (LoRA folded into W2eff)
  prep_w1_kernel<<<(D * D) / (256 * 8), 256, 0, stream>>>(W1, W1b);
  prep_w2_kernel<<<(D * D) / (256 * 4), 256, 0, stream>>>(W2, A, Bl, W2b);

  dim3 grid(D / BN, M_TOT / BM);   // 16 x 128
  // GEMM1: h = gelu(x @ W1^T + b1), x converted fp32->bf16 while staging
  gemm_bf16_wmma<true, true><<<grid, 256, 0, stream>>>(x, W1b, b1, h);
  // GEMM2: out = h @ W2eff^T + b2  (fp32 output)
  gemm_bf16_wmma<false, false><<<grid, 256, 0, stream>>>(h, W2b, b2, (float*)d_out);
}
